// MagicGAT_44959717655161
// MI455X (gfx1250) — matrix operand order repaired and checked by reading, not compiled, not run on hardware
//
#include <hip/hip_runtime.h>
#include <hip/hip_bf16.h>
#include <stdint.h>

// Problem constants (match reference)
#define N_NODES 100000
#define N_EDGES 1600000
#define HC      128
#define NHEAD   4
#define CDIM    32
#define GAT_SLOPE 0.2f

typedef __attribute__((ext_vector_type(16))) __bf16        v16bf;
typedef __attribute__((ext_vector_type(8)))  float         v8f;
typedef __attribute__((ext_vector_type(4)))  unsigned int  uint32x4;
typedef __attribute__((ext_vector_type(4)))  int           int32x4;
typedef __attribute__((ext_vector_type(8)))  int           int32x8;

__device__ __forceinline__ unsigned short f2bfbits(float f) {
    union { float f; unsigned u; } v; v.f = f;
    unsigned u = v.u;
    unsigned r = (u + 0x7FFFu + ((u >> 16) & 1u)) >> 16;   // round-to-nearest-even
    return (unsigned short)r;
}

// ---------------------------------------------------------------------------
// Tensor Data Mover: async-load a [rows x 128] bf16 tile (row stride 128 elems)
// from global into LDS at lds_byte_off, padding +16B per 256B row so the LDS
// tile has a 272B (136-half) row stride (bank-conflict-free ds_load_b128).
// D# bitfields per CDNA5 ISA.
// ---------------------------------------------------------------------------
__device__ __forceinline__ void tdm_load_tile_bf16(unsigned lds_byte_off,
                                                   const unsigned short* gsrc,
                                                   int rows) {
    unsigned long long ga = (unsigned long long)(uintptr_t)gsrc;
    uint32x4 g0;
    g0[0] = 1u;                                       // count=1 (valid user D#)
    g0[1] = lds_byte_off;                             // lds_addr (bytes)
    g0[2] = (unsigned)(ga & 0xFFFFFFFFu);             // global_addr[31:0]
    g0[3] = (unsigned)((ga >> 32) & 0x01FFFFFFu)      // global_addr[56:32]
          | (2u << 30);                               // type = 2 ("image")
    int32x8 g1;
    g1[0] = (1 << 16)                                 // data_size = 2 bytes
          | (1 << 20)                                 // pad_enable
          | (5 << 22)                                 // pad_interval: 64 dwords (=256B row)
          | (3 << 25);                                // pad_amount: 4 dwords (=16B)
    g1[1] = (128 << 16);                              // tensor_dim0[15:0] = 128
    g1[2] = (rows << 16);                             // tensor_dim0[31:16]=0 | tensor_dim1 lo
    g1[3] = (128 << 16);                              // tensor_dim1 hi = 0 | tile_dim0 = 128
    g1[4] = rows;                                     // tile_dim1 = rows, tile_dim2 = 0
    g1[5] = 128;                                      // tensor_dim0_stride (elems) low 32
    g1[6] = 0;
    g1[7] = 0;
    int32x4 z4 = {0, 0, 0, 0};
#if __has_include(<hip/amd_detail/amd_gfx1250_TDM.h>)
    int32x8 z8 = {0, 0, 0, 0, 0, 0, 0, 0};
    __builtin_amdgcn_tensor_load_to_lds(g0, g1, z4, z4, z8, 0);
#else
    __builtin_amdgcn_tensor_load_to_lds(g0, g1, z4, z4, 0);
#endif
}

// Bit-cast helpers for fragment assembly
union FragB { uint4 q[2]; v16bf v; };

// A fragment from a padded LDS tile (stride 136 halves): contiguous-k A order, same as packed B.
// Lane r=lane&15, hi=lane>>4: elements 0-15 = K [ak, ak+16), ak = kbase + hi*16
// -> two aligned 16-byte LDS reads.
__device__ __forceinline__ v16bf load_a_frag(const unsigned short* At,
                                             int r, int hi, int kbase) {
    const int ak = kbase + hi * 16;
    FragB f;
    f.q[0] = *(const uint4*)&At[r * 136 + ak];
    f.q[1] = *(const uint4*)&At[r * 136 + ak + 8];
    return f.v;
}

// B fragment from fragment-packed weights: 32 contiguous bytes per lane.
__device__ __forceinline__ v16bf load_b_frag(const unsigned short* Wp,
                                             int tile, int KC, int kk, int lane) {
    const size_t base = ((size_t)(tile * KC + kk) * 32 + lane) * 16;
    FragB f;
    f.q[0] = *(const uint4*)&Wp[base];
    f.q[1] = *(const uint4*)&Wp[base + 8];
    return f.v;
}

// ---------------------------------------------------------------------------
// f32 -> bf16 (activations, row-major unchanged)
// ---------------------------------------------------------------------------
__global__ void cvt_act_kernel(const float* __restrict__ in,
                               unsigned short* __restrict__ out, int n) {
    int t = blockIdx.x * blockDim.x + threadIdx.x;
    if (t < n) out[t] = f2bfbits(in[t]);
}

// ---------------------------------------------------------------------------
// Pack weight [K, Ncols] (f32, row = k) into WMMA B-fragment order (bf16):
// out[((tile*KC + kc)*32 + lane)*16 + j] = W[(kc*32 + (lane>>4)*16 + j)*Ncols
//                                            + tile*16 + (lane&15)]
// ---------------------------------------------------------------------------
__global__ void pack_w_kernel(const float* __restrict__ in,
                              unsigned short* __restrict__ out,
                              int K, int Ncols) {
    int t = blockIdx.x * blockDim.x + threadIdx.x;
    if (t >= K * Ncols) return;
    const int KC   = K >> 5;
    int j    = t & 15;
    int lane = (t >> 4) & 31;
    int rem  = t >> 9;
    int kc   = rem % KC;
    int tile = rem / KC;
    int n = tile * 16 + (lane & 15);
    int k = kc * 32 + (lane >> 4) * 16 + j;
    out[t] = f2bfbits(in[k * Ncols + n]);
}

// ---------------------------------------------------------------------------
// xh = A @ W (bf16 WMMA, f32 accum) + fused attention logits es/ed.
// A tile arrives via TDM; B from fragment-packed weights.
// Block = 256 threads (8 waves); wave w owns output cols [16w, 16w+16).
// ---------------------------------------------------------------------------
__global__ __launch_bounds__(256) void gemm_att_kernel(
    const unsigned short* __restrict__ Ab,    // [N,128] bf16 activations
    const unsigned short* __restrict__ Wp,    // packed [8][4][32][16] bf16
    const float* __restrict__ a_s,            // [128]
    const float* __restrict__ a_d,            // [128]
    float* __restrict__ xh,                   // [N,128]
    float* __restrict__ es,                   // [N,4]
    float* __restrict__ ed)                   // [N,4]
{
    __shared__ unsigned short At[16 * 136];   // TDM-padded tile (272B row stride)
    __shared__ float es_l[64];
    __shared__ float ed_l[64];

    const int tid  = threadIdx.x;
    const int row0 = blockIdx.x * 16;

    __builtin_prefetch(Wp, 0, 3);             // global_prefetch_b8

    if (tid < 32) {                           // wave 0 issues the tensor DMA
        tdm_load_tile_bf16((unsigned)(uintptr_t)&At[0],
                           Ab + (size_t)row0 * HC, 16);
        __builtin_amdgcn_s_wait_tensorcnt(0);
    }
    if (tid < 64) { es_l[tid] = 0.f; ed_l[tid] = 0.f; }
    __syncthreads();

    const int wave = tid >> 5;
    const int lane = tid & 31;
    const int r    = lane & 15;
    const int hi   = lane >> 4;

    v8f acc = {};
    #pragma unroll
    for (int kk = 0; kk < 4; ++kk) {
        v16bf a = load_a_frag(At, r, hi, kk * 32);
        v16bf b = load_b_frag(Wp, wave, 4, kk, lane);
        acc = __builtin_amdgcn_wmma_f32_16x16x32_bf16(false, a, false, b,
                                                      (short)0, acc, false, false);
    }

    const int   col  = wave * 16 + r;
    const int   head = col >> 5;
    const float asv  = a_s[col];
    const float adv  = a_d[col];
    #pragma unroll
    for (int i = 0; i < 8; ++i) {
        const int m = hi * 8 + i;             // C/D layout: M = hi*8 + vgpr index
        xh[(size_t)(row0 + m) * HC + col] = acc[i];
        atomicAdd(&es_l[m * 4 + head], acc[i] * asv);
        atomicAdd(&ed_l[m * 4 + head], acc[i] * adv);
    }
    __syncthreads();
    if (tid < 64) {
        es[(size_t)row0 * 4 + tid] = es_l[tid];
        ed[(size_t)row0 * 4 + tid] = ed_l[tid];
    }
}

// ---------------------------------------------------------------------------
// out = A @ W + bias1 + bias2   (h2 = h1@llW + llb + b2)
// ---------------------------------------------------------------------------
__global__ __launch_bounds__(256) void gemm_plain_kernel(
    const unsigned short* __restrict__ Ab,
    const unsigned short* __restrict__ Wp,
    const float* __restrict__ bias1,
    const float* __restrict__ bias2,
    float* __restrict__ out)
{
    __shared__ unsigned short At[16 * 136];
    const int tid  = threadIdx.x;
    const int row0 = blockIdx.x * 16;

    __builtin_prefetch(Wp, 0, 3);

    if (tid < 32) {
        tdm_load_tile_bf16((unsigned)(uintptr_t)&At[0],
                           Ab + (size_t)row0 * HC, 16);
        __builtin_amdgcn_s_wait_tensorcnt(0);
    }
    __syncthreads();

    const int wave = tid >> 5;
    const int lane = tid & 31;
    const int r    = lane & 15;
    const int hi   = lane >> 4;

    v8f acc = {};
    #pragma unroll
    for (int kk = 0; kk < 4; ++kk) {
        v16bf a = load_a_frag(At, r, hi, kk * 32);
        v16bf b = load_b_frag(Wp, wave, 4, kk, lane);
        acc = __builtin_amdgcn_wmma_f32_16x16x32_bf16(false, a, false, b,
                                                      (short)0, acc, false, false);
    }

    const int col = wave * 16 + r;
    const float bsum = bias1[col] + bias2[col];
    #pragma unroll
    for (int i = 0; i < 8; ++i) {
        const int m = hi * 8 + i;
        out[(size_t)(row0 + m) * HC + col] = acc[i] + bsum;
    }
}

// ---------------------------------------------------------------------------
// out[N,32] = concat(h0,h1,h2)[N,384] @ opW[384,32] + opb
// Block = 64 threads (2 waves); three TDM tile loads into one LDS staging area.
// ---------------------------------------------------------------------------
__global__ __launch_bounds__(64) void out_proj_kernel(
    const unsigned short* __restrict__ h0b,
    const unsigned short* __restrict__ h1b,
    const unsigned short* __restrict__ h2b,
    const unsigned short* __restrict__ Wp,    // packed [2][12][32][16]
    const float* __restrict__ bias,           // [32]
    float* __restrict__ out)                  // [N,32]
{
    __shared__ unsigned short At[3 * 16 * 136];  // 3 padded sub-tiles

    const int tid  = threadIdx.x;
    const int row0 = blockIdx.x * 16;

    __builtin_prefetch(Wp, 0, 3);

    if (tid < 32) {
        const unsigned base = (unsigned)(uintptr_t)&At[0];
        tdm_load_tile_bf16(base,            h0b + (size_t)row0 * HC, 16);
        tdm_load_tile_bf16(base + 4352,     h1b + (size_t)row0 * HC, 16);
        tdm_load_tile_bf16(base + 2 * 4352, h2b + (size_t)row0 * HC, 16);
        __builtin_amdgcn_s_wait_tensorcnt(0);
    }
    __syncthreads();

    const int wave = tid >> 5;
    const int lane = tid & 31;
    const int r    = lane & 15;
    const int hi   = lane >> 4;

    v8f acc = {};
    #pragma unroll
    for (int kk = 0; kk < 12; ++kk) {
        const unsigned short* sub = &At[(kk >> 2) * 16 * 136]; // buffer = kk/4
        v16bf a = load_a_frag(sub, r, hi, (kk & 3) * 32);
        v16bf b = load_b_frag(Wp, wave, 12, kk, lane);
        acc = __builtin_amdgcn_wmma_f32_16x16x32_bf16(false, a, false, b,
                                                      (short)0, acc, false, false);
    }

    const int col = wave * 16 + r;
    const float bv = bias[col];
    #pragma unroll
    for (int i = 0; i < 8; ++i) {
        const int m = hi * 8 + i;
        out[(size_t)(row0 + m) * 32 + col] = acc[i] + bv;
    }
}

// ---------------------------------------------------------------------------
// Edge phase kernels
// ---------------------------------------------------------------------------
__global__ void edge_state_init_kernel(float* __restrict__ emax,
                                       float* __restrict__ denom, int n) {
    int t = blockIdx.x * blockDim.x + threadIdx.x;
    if (t < n) {
        ((unsigned*)emax)[t] = 0xFF800000u;   // -inf
        denom[t] = 0.f;
    }
}

__global__ void node_init_kernel(const float* __restrict__ res,
                                 const float* __restrict__ bias,
                                 float* __restrict__ out, int n) {
    int t = blockIdx.x * blockDim.x + threadIdx.x;
    if (t < n) {
        float v = bias[t & (HC - 1)];
        if (res) v += res[t];
        out[t] = v;
    }
}

__device__ __forceinline__ float lrelu(float v) {
    return v > 0.f ? v : GAT_SLOPE * v;
}

__device__ __forceinline__ void atomicMaxF32(float* addr, float val) {
    int iv = __float_as_int(val);
    if (iv >= 0) atomicMax((int*)addr, iv);
    else         atomicMin((unsigned int*)addr, (unsigned int)iv);
}

// one thread per edge: all 4 heads via float4
__global__ void edge_max_kernel(const int* __restrict__ src,
                                const int* __restrict__ dst,
                                const float* __restrict__ es,
                                const float* __restrict__ ed,
                                float* __restrict__ emax) {
    int e = blockIdx.x * blockDim.x + threadIdx.x;
    if (e >= N_EDGES) return;
    int s = src[e], d = dst[e];
    float4 a = *(const float4*)&es[(size_t)s * 4];
    float4 b = *(const float4*)&ed[(size_t)d * 4];
    float* m = (float*)&emax[(size_t)d * 4];
    atomicMaxF32(m + 0, lrelu(a.x + b.x));
    atomicMaxF32(m + 1, lrelu(a.y + b.y));
    atomicMaxF32(m + 2, lrelu(a.z + b.z));
    atomicMaxF32(m + 3, lrelu(a.w + b.w));
}

__global__ void edge_exp_kernel(const int* __restrict__ src,
                                const int* __restrict__ dst,
                                const float* __restrict__ es,
                                const float* __restrict__ ed,
                                const float* __restrict__ emax,
                                float* __restrict__ ee,
                                float* __restrict__ denom) {
    int e = blockIdx.x * blockDim.x + threadIdx.x;
    if (e >= N_EDGES) return;
    int s = src[e], d = dst[e];
    float4 a = *(const float4*)&es[(size_t)s * 4];
    float4 b = *(const float4*)&ed[(size_t)d * 4];
    float4 m = *(const float4*)&emax[(size_t)d * 4];
    float4 x;
    x.x = __expf(lrelu(a.x + b.x) - m.x);
    x.y = __expf(lrelu(a.y + b.y) - m.y);
    x.z = __expf(lrelu(a.z + b.z) - m.z);
    x.w = __expf(lrelu(a.w + b.w) - m.w);
    *(float4*)&ee[(size_t)e * 4] = x;
    float* dn = (float*)&denom[(size_t)d * 4];
    atomicAdd(dn + 0, x.x);
    atomicAdd(dn + 1, x.y);
    atomicAdd(dn + 2, x.z);
    atomicAdd(dn + 3, x.w);
}

// out[dst] += alpha * xh[src]; 32 threads per edge, 4 channels each
__global__ void aggregate_kernel(const int* __restrict__ src,
                                 const int* __restrict__ dst,
                                 const float* __restrict__ xh,
                                 const float* __restrict__ ee,
                                 const float* __restrict__ denom,
                                 float* __restrict__ out) {
    long long t = (long long)blockIdx.x * blockDim.x + threadIdx.x;
    if (t >= (long long)N_EDGES * 32) return;
    int e = (int)(t >> 5);
    int q = (int)(t & 31);
    int s = src[e], d = dst[e];
    int head = q >> 3;
    float alpha = ee[(size_t)e * 4 + head] / denom[(size_t)d * 4 + head];
    const float4 xv = *(const float4*)&xh[(size_t)s * HC + q * 4];
    float* o = &out[(size_t)d * HC + q * 4];
    atomicAdd(o + 0, xv.x * alpha);
    atomicAdd(o + 1, xv.y * alpha);
    atomicAdd(o + 2, xv.z * alpha);
    atomicAdd(o + 3, xv.w * alpha);
}

// ---------------------------------------------------------------------------
// Host orchestration
// ---------------------------------------------------------------------------
extern "C" void kernel_launch(void* const* d_in, const int* in_sizes, int n_in,
                              void* d_out, int out_size, void* d_ws, size_t ws_size,
                              hipStream_t stream) {
    (void)in_sizes; (void)n_in; (void)out_size; (void)ws_size;

    const float* x   = (const float*)d_in[0];
    const int*   ei  = (const int*)d_in[1];
    const int*   src = ei;
    const int*   dst = ei + N_EDGES;
    const float* W[3]  = {(const float*)d_in[2],  (const float*)d_in[6],  (const float*)d_in[10]};
    const float* AS[3] = {(const float*)d_in[3],  (const float*)d_in[7],  (const float*)d_in[11]};
    const float* AD[3] = {(const float*)d_in[4],  (const float*)d_in[8],  (const float*)d_in[12]};
    const float* B[3]  = {(const float*)d_in[5],  (const float*)d_in[9],  (const float*)d_in[13]};
    const float* llW = (const float*)d_in[14];
    const float* llb = (const float*)d_in[15];
    const float* opW = (const float*)d_in[16];
    const float* opb = (const float*)d_in[17];
    float* out = (float*)d_out;

    // Workspace carve-up
    float* ws = (float*)d_ws;
    const size_t nh = (size_t)N_NODES * HC;
    float* h0    = ws;  ws += nh;
    float* h1    = ws;  ws += nh;
    float* h2    = ws;  ws += nh;
    float* xh    = ws;  ws += nh;
    float* es    = ws;  ws += (size_t)N_NODES * 4;
    float* ed    = ws;  ws += (size_t)N_NODES * 4;
    float* emax  = ws;  ws += (size_t)N_NODES * 4;
    float* denom = ws;  ws += (size_t)N_NODES * 4;
    float* ee    = ws;  ws += (size_t)N_EDGES * 4;
    unsigned short* bfp = (unsigned short*)ws;
    unsigned short* xb  = bfp;  bfp += nh;       // bf16 activations
    unsigned short* h0b = bfp;  bfp += nh;
    unsigned short* h1b = bfp;  bfp += nh;
    unsigned short* h2b = bfp;  bfp += nh;
    unsigned short* Wp[3];
    Wp[0] = bfp;                bfp += HC * HC;  // fragment-packed weights
    Wp[1] = bfp;                bfp += HC * HC;
    Wp[2] = bfp;                bfp += HC * HC;
    unsigned short* llWp = bfp; bfp += HC * HC;
    unsigned short* opWp = bfp; bfp += 3 * HC * CDIM;

    const int THREADS = 256;
    const int nWElem   = HC * HC;
    const int rowTiles = N_NODES / 16;                       // 6250
    const int edgeBlocks  = (N_EDGES + THREADS - 1) / THREADS;
    const int aggBlocks   = (int)(((long long)N_EDGES * 32 + THREADS - 1) / THREADS);
    const int nodeBlocks  = (int)((nh + THREADS - 1) / THREADS);
    const int stateBlocks = (N_NODES * 4 + THREADS - 1) / THREADS;

    // Pack weights into WMMA B-fragment layout (bf16)
    for (int i = 0; i < 3; ++i)
        pack_w_kernel<<<(nWElem + THREADS - 1) / THREADS, THREADS, 0, stream>>>(W[i], Wp[i], HC, HC);
    pack_w_kernel<<<(nWElem + THREADS - 1) / THREADS, THREADS, 0, stream>>>(llW, llWp, HC, HC);
    pack_w_kernel<<<(3 * HC * CDIM + THREADS - 1) / THREADS, THREADS, 0, stream>>>(opW, opWp, 3 * HC, CDIM);

    // bf16 copy of input activations
    cvt_act_kernel<<<nodeBlocks, THREADS, 0, stream>>>(x, xb, (int)nh);

    const unsigned short* layer_in_b[3] = {xb, h0b, h1b};
    float*          layer_out[3]  = {h0, h1, h2};
    unsigned short* layer_outb[3] = {h0b, h1b, h2b};
    const float*    layer_res[3]  = {nullptr, h0, nullptr};

    for (int L = 0; L < 3; ++L) {
        // xh = in @ W (+ fused attention logits), A tile via TDM
        gemm_att_kernel<<<rowTiles, THREADS, 0, stream>>>(
            layer_in_b[L], Wp[L], AS[L], AD[L], xh, es, ed);

        edge_state_init_kernel<<<stateBlocks, THREADS, 0, stream>>>(emax, denom, N_NODES * 4);
        if (L == 2) {
            gemm_plain_kernel<<<rowTiles, THREADS, 0, stream>>>(h1b, llWp, llb, B[2], h2);
        } else {
            node_init_kernel<<<nodeBlocks, THREADS, 0, stream>>>(layer_res[L], B[L], layer_out[L], (int)nh);
        }

        edge_max_kernel<<<edgeBlocks, THREADS, 0, stream>>>(src, dst, es, ed, emax);
        edge_exp_kernel<<<edgeBlocks, THREADS, 0, stream>>>(src, dst, es, ed, emax, ee, denom);
        aggregate_kernel<<<aggBlocks, THREADS, 0, stream>>>(src, dst, xh, ee, denom, layer_out[L]);

        // bf16 copy for the next layer's GEMM / final projection
        cvt_act_kernel<<<nodeBlocks, THREADS, 0, stream>>>(layer_out[L], layer_outb[L], (int)nh);
    }

    // final projection: concat(h0,h1,h2) @ opW + opb
    out_proj_kernel<<<rowTiles, 64, 0, stream>>>(h0b, h1b, h2b, opWp, opb, out);
}
